// SAGPoolGNN_26036091748792
// MI455X (gfx1250) — matrix operand order repaired and checked by reading, not compiled, not run on hardware
//
#include <hip/hip_runtime.h>
#include <hip/hip_bf16.h>
#include <math.h>

#define D 128
#define B_GRAPHS 128
#define N0 2048
#define NT0 (B_GRAPHS * N0)      /* 262144 nodes at layer 0 */
#define E_EDGES (NT0 * 8)        /* 2097152 edges (static, masked) */
#define NUM_CLASS 40

typedef __attribute__((ext_vector_type(16))) __bf16 v16bf;
typedef __attribute__((ext_vector_type(8)))  float  v8f;

__device__ __forceinline__ __bf16 f2bf(float f) { return (__bf16)f; }

__device__ __forceinline__ unsigned f2key(float f) {
  unsigned u = __float_as_uint(f);
  return (u & 0x80000000u) ? ~u : (u | 0x80000000u);
}

// ---------------------------------------------------------------------------
// Fill helpers
// ---------------------------------------------------------------------------
__global__ void fill_f32_kernel(float* p, float v, int n) {
  int t = blockIdx.x * blockDim.x + threadIdx.x;
  if (t < n) p[t] = v;
}
__global__ void fill_i32_kernel(int* p, int v, int n) {
  int t = blockIdx.x * blockDim.x + threadIdx.x;
  if (t < n) p[t] = v;
}

// ---------------------------------------------------------------------------
// Embedding: h[n,d] = emb_type[x[n],d] + emb_depth[depth[n],d]
// ---------------------------------------------------------------------------
__global__ void embed_kernel(const int* __restrict__ x, const int* __restrict__ dep,
                             const float* __restrict__ et, const float* __restrict__ ed,
                             float* __restrict__ h) {
  int t = blockIdx.x * blockDim.x + threadIdx.x;
  if (t >= NT0 * D) return;
  int node = t >> 7, d = t & (D - 1);
  h[t] = et[x[node] * D + d] + ed[dep[node] * D + d];
}

// ---------------------------------------------------------------------------
// Copy edges into mutable workspace, emask = 1
// ---------------------------------------------------------------------------
__global__ void init_edges_kernel(const int* __restrict__ ei, int* __restrict__ s,
                                  int* __restrict__ dd, float* __restrict__ em) {
  int t = blockIdx.x * blockDim.x + threadIdx.x;
  if (t >= E_EDGES) return;
  s[t] = ei[t];
  dd[t] = ei[E_EDGES + t];
  em[t] = 1.0f;
}

// ---------------------------------------------------------------------------
// Pack conv_W (3 x 128 x 128 f32, row-major K x N) into per-lane bf16
// B-fragments matching the V_WMMA 16-bit B layout.
// fragment index = (kc*8 + j)*32 + lane, element e:
//   K = kc*32 + 8*(lane>=16) + (e<8 ? e : e+8) ;  N = j*16 + (lane&15)
// ---------------------------------------------------------------------------
__global__ void pack_w_kernel(const float* __restrict__ convW, __bf16* __restrict__ wp) {
  int t = blockIdx.x * blockDim.x + threadIdx.x;
  if (t >= 3 * 4 * 8 * 32 * 16) return;
  int e = t & 15;
  int lane = (t >> 4) & 31;
  int j = (t >> 9) & 7;
  int kc = (t >> 12) & 3;
  int l = t >> 14;
  int K = kc * 32 + ((lane >> 4) ? 8 : 0) + (e < 8 ? e : e + 8);
  int N = j * 16 + (lane & 15);
  wp[t] = f2bf(convW[(size_t)l * D * D + K * D + N]);
}

// ---------------------------------------------------------------------------
// hout[M,128] = hin[M,128] @ W[128,128] via v_wmma_f32_16x16x32_bf16.
// 256 threads = 8 waves; each wave owns a 16x128 strip (8 N-tiles), block
// covers 128 rows. K loop: 4 chunks of 32.
// ---------------------------------------------------------------------------
__global__ void gemm_bf16_kernel(const float* __restrict__ hin,
                                 const __bf16* __restrict__ wpack,
                                 float* __restrict__ hout, int M) {
  const int lane = threadIdx.x & 31;
  const int wave = threadIdx.x >> 5;
  const int row0 = blockIdx.x * 128 + wave * 16;
  if (row0 >= M) return;
  const int half = lane >> 4;            // 0 or 1
  const int r = row0 + (lane & 15);

  v8f acc[8] = {};                        // 16x16 f32 C tiles, 8 N-tiles
#pragma unroll
  for (int kc = 0; kc < 4; ++kc) {
    // ---- A fragment: lane<16 -> K in {kb..kb+7, kb+16..kb+23}, kb=kc*32;
    //      lane>=16 -> kb += 8  (ISA 16-bit A 16x32 layout)
    const int kb = kc * 32 + half * 8;
    const float* rowp = hin + (size_t)r * D + kb;
    float4 f0 = *(const float4*)(rowp);
    float4 f1 = *(const float4*)(rowp + 4);
    float4 f2 = *(const float4*)(rowp + 16);
    float4 f3 = *(const float4*)(rowp + 20);
    v16bf a;
    a[0]  = f2bf(f0.x); a[1]  = f2bf(f0.y); a[2]  = f2bf(f0.z); a[3]  = f2bf(f0.w);
    a[4]  = f2bf(f1.x); a[5]  = f2bf(f1.y); a[6]  = f2bf(f1.z); a[7]  = f2bf(f1.w);
    a[8]  = f2bf(f2.x); a[9]  = f2bf(f2.y); a[10] = f2bf(f2.z); a[11] = f2bf(f2.w);
    a[12] = f2bf(f3.x); a[13] = f2bf(f3.y); a[14] = f2bf(f3.z); a[15] = f2bf(f3.w);
#pragma unroll
    for (int j = 0; j < 8; ++j) {
      v16bf b = *(const v16bf*)(wpack + (size_t)(((kc * 8 + j) * 32) + lane) * 16);
      acc[j] = __builtin_amdgcn_wmma_f32_16x16x32_bf16(
          false, a, false, b, (short)0, acc[j], false, false);
    }
  }
  // ---- store C: VGPR g -> M = row0 + g + 8*half, N = j*16 + (lane&15)
  const int mrow = row0 + half * 8;
  const int ncol = lane & 15;
#pragma unroll
  for (int j = 0; j < 8; ++j)
#pragma unroll
    for (int g = 0; g < 8; ++g)
      hout[(size_t)(mrow + g) * D + j * 16 + ncol] = acc[j][g];
}

// ---------------------------------------------------------------------------
// Degree: deg pre-filled with 1.0 (self loop); add emask at dst.
// ---------------------------------------------------------------------------
__global__ void deg_scatter_kernel(const int* __restrict__ dst,
                                   const float* __restrict__ em,
                                   float* __restrict__ deg) {
  int t = blockIdx.x * blockDim.x + threadIdx.x;
  if (t >= E_EDGES) return;
  float m = em[t];
  if (m != 0.0f) atomicAdd(&deg[dst[t]], m);
}
__global__ void rsqrt_kernel(float* __restrict__ deg, int n) {
  int t = blockIdx.x * blockDim.x + threadIdx.x;
  if (t < n) deg[t] = rsqrtf(deg[t]);
}

// ---------------------------------------------------------------------------
// agg[dst] += hw[src] * dinv[src]*dinv[dst]*emask   (one wave32 per edge,
// 4 f32 atomics per lane). h fits in the 192MB L2 so this runs out of L2.
// ---------------------------------------------------------------------------
__global__ void gcn_scatter_kernel(const float* __restrict__ hw,
                                   const int* __restrict__ src,
                                   const int* __restrict__ dst,
                                   const float* __restrict__ em,
                                   const float* __restrict__ dinv,
                                   float* __restrict__ agg) {
  int wid = (blockIdx.x * blockDim.x + threadIdx.x) >> 5;
  int lane = threadIdx.x & 31;
  if (wid >= E_EDGES) return;
  int nxt = wid + 64;
  if (nxt < E_EDGES)  // warm L2/L0 for an upcoming source row (gfx1250 prefetch)
    __builtin_prefetch(hw + (size_t)src[nxt] * D + lane * 4, 0, 1);
  int s = src[wid], d = dst[wid];
  float nrm = dinv[s] * dinv[d] * em[wid];
  if (nrm == 0.0f) return;
  float4 v = *(const float4*)(hw + (size_t)s * D + lane * 4);
  float* ap = agg + (size_t)d * D + lane * 4;
  atomicAdd(ap + 0, v.x * nrm);
  atomicAdd(ap + 1, v.y * nrm);
  atomicAdd(ap + 2, v.z * nrm);
  atomicAdd(ap + 3, v.w * nrm);
}

// ---------------------------------------------------------------------------
// h2 = relu(agg + hw*dinv^2 + b)   (in place over agg)
// ---------------------------------------------------------------------------
__global__ void combine_kernel(float* __restrict__ agg, const float* __restrict__ hw,
                               const float* __restrict__ dinv,
                               const float* __restrict__ bias, int nt) {
  int t = blockIdx.x * blockDim.x + threadIdx.x;
  if (t >= nt * D) return;
  int node = t >> 7, d = t & (D - 1);
  float di = dinv[node];
  float v = agg[t] + hw[t] * di * di + bias[d];
  agg[t] = v > 0.0f ? v : 0.0f;
}

// ---------------------------------------------------------------------------
// Per-node score pieces: sc[i] = dot(h[i],Wr)+b ; tvec[i] = dot(h[i],Wn)
// One wave per node, wave32 shuffle reduction.
// ---------------------------------------------------------------------------
__global__ void score_node_kernel(const float* __restrict__ h,
                                  const float* __restrict__ Wr,
                                  const float* __restrict__ Wn,
                                  const float* __restrict__ bptr,
                                  float* __restrict__ sc, float* __restrict__ tvec,
                                  int nt) {
  int wid = (blockIdx.x * blockDim.x + threadIdx.x) >> 5;
  int lane = threadIdx.x & 31;
  if (wid >= nt) return;
  float4 hv = *(const float4*)(h + (size_t)wid * D + lane * 4);
  float4 wr = *(const float4*)(Wr + lane * 4);
  float4 wn = *(const float4*)(Wn + lane * 4);
  float sr = hv.x * wr.x + hv.y * wr.y + hv.z * wr.z + hv.w * wr.w;
  float sn = hv.x * wn.x + hv.y * wn.y + hv.z * wn.z + hv.w * wn.w;
#pragma unroll
  for (int off = 16; off > 0; off >>= 1) {
    sr += __shfl_xor(sr, off, 32);
    sn += __shfl_xor(sn, off, 32);
  }
  if (lane == 0) {
    sc[wid] = sr + bptr[0];
    tvec[wid] = sn;
  }
}

__global__ void score_scatter_kernel(const int* __restrict__ src,
                                     const int* __restrict__ dst,
                                     const float* __restrict__ em,
                                     const float* __restrict__ tvec,
                                     float* __restrict__ sc) {
  int t = blockIdx.x * blockDim.x + threadIdx.x;
  if (t >= E_EDGES) return;
  float m = em[t];
  if (m != 0.0f) atomicAdd(&sc[dst[t]], tvec[src[t]] * m);
}

// ---------------------------------------------------------------------------
// Per-graph top-k via bitonic sort (descending) in LDS. Readouts & message
// passing are permutation-invariant, so only the top-k *set* matters.
// ---------------------------------------------------------------------------
__global__ void topk_kernel(const float* __restrict__ sc, int n, int k,
                            int* __restrict__ kept, int* __restrict__ newpos) {
  __shared__ unsigned skey[N0];
  __shared__ int sidx[N0];
  int g = blockIdx.x, tid = threadIdx.x;
  for (int i = tid; i < n; i += blockDim.x) {
    skey[i] = f2key(sc[(size_t)g * n + i]);
    sidx[i] = i;
  }
  __syncthreads();
  for (int ksz = 2; ksz <= n; ksz <<= 1) {
    for (int st = ksz >> 1; st > 0; st >>= 1) {
      for (int i = tid; i < n; i += blockDim.x) {
        int j = i ^ st;
        if (j > i) {
          bool up = (i & ksz) != 0;  // overall descending order
          unsigned a = skey[i], b = skey[j];
          if (up ? (a > b) : (a < b)) {
            skey[i] = b; skey[j] = a;
            int ti = sidx[i]; sidx[i] = sidx[j]; sidx[j] = ti;
          }
        }
      }
      __syncthreads();
    }
  }
  for (int r = tid; r < k; r += blockDim.x) {
    int old = g * n + sidx[r];
    kept[g * k + r] = old;
    newpos[old] = g * k + r;
  }
}

// ---------------------------------------------------------------------------
// h_next[m] = h2[kept[m]] * tanh(score[kept[m]])
// ---------------------------------------------------------------------------
__global__ void pool_gather_kernel(const float* __restrict__ h2,
                                   const int* __restrict__ kept,
                                   const float* __restrict__ sc,
                                   float* __restrict__ hnext, int bk) {
  int t = blockIdx.x * blockDim.x + threadIdx.x;
  if (t >= bk * D) return;
  int m = t >> 7, d = t & (D - 1);
  int old = kept[m];
  hnext[t] = h2[(size_t)old * D + d] * tanhf(sc[old]);
}

__global__ void remap_kernel(const int* __restrict__ newpos, int* __restrict__ src,
                             int* __restrict__ dst, float* __restrict__ em) {
  int t = blockIdx.x * blockDim.x + threadIdx.x;
  if (t >= E_EDGES) return;
  int ns = newpos[src[t]], nd = newpos[dst[t]];
  bool valid = (ns >= 0) && (nd >= 0) && (em[t] > 0.0f);
  src[t] = valid ? ns : 0;
  dst[t] = valid ? nd : 0;
  em[t] = valid ? 1.0f : 0.0f;
}

// ---------------------------------------------------------------------------
// Readout: gsum[g, 0:128] += max over nodes, gsum[g, 128:256] += mean.
// ---------------------------------------------------------------------------
__global__ void readout_kernel(const float* __restrict__ h, int n,
                               float* __restrict__ gsum) {
  int g = blockIdx.x, t = threadIdx.x;
  const float* base = h + (size_t)g * n * D;
  if (t < D) {
    float m = -3.0e38f;
    for (int j = 0; j < n; ++j) m = fmaxf(m, base[(size_t)j * D + t]);
    gsum[g * 2 * D + t] += m;
  } else {
    int c = t - D;
    float s = 0.0f;
    for (int j = 0; j < n; ++j) s += base[(size_t)j * D + c];
    gsum[g * 2 * D + D + c] += s / (float)n;
  }
}

// ---------------------------------------------------------------------------
// Final MLP head: 256 -> 128 relu -> 128 relu -> 40. One block per graph.
// ---------------------------------------------------------------------------
__global__ void mlp_kernel(const float* __restrict__ gsum,
                           const float* __restrict__ l1W, const float* __restrict__ l1b,
                           const float* __restrict__ l2W, const float* __restrict__ l2b,
                           const float* __restrict__ hW, const float* __restrict__ hb,
                           float* __restrict__ out) {
  __shared__ float gv[2 * D];
  __shared__ float o1[D];
  __shared__ float o2[D];
  int g = blockIdx.x, t = threadIdx.x;
  gv[t] = gsum[g * 2 * D + t];
  gv[t + D] = gsum[g * 2 * D + D + t];
  __syncthreads();
  float s = l1b[t];
  for (int k = 0; k < 2 * D; ++k) s += gv[k] * l1W[k * D + t];
  o1[t] = s > 0.0f ? s : 0.0f;
  __syncthreads();
  s = l2b[t];
  for (int k = 0; k < D; ++k) s += o1[k] * l2W[k * D + t];
  o2[t] = s > 0.0f ? s : 0.0f;
  __syncthreads();
  if (t < NUM_CLASS) {
    s = hb[t];
    for (int k = 0; k < D; ++k) s += o2[k] * hW[k * NUM_CLASS + t];
    out[g * NUM_CLASS + t] = s;
  }
}

// ---------------------------------------------------------------------------
extern "C" void kernel_launch(void* const* d_in, const int* in_sizes, int n_in,
                              void* d_out, int out_size, void* d_ws, size_t ws_size,
                              hipStream_t stream) {
  const int* x = (const int*)d_in[0];
  const int* depth = (const int*)d_in[1];
  const int* edge_index = (const int*)d_in[2];
  /* d_in[3] = batch (implicit from B/N0) */
  const float* emb_type = (const float*)d_in[4];
  const float* emb_depth = (const float*)d_in[5];
  const float* conv_W = (const float*)d_in[6];
  const float* conv_b = (const float*)d_in[7];
  const float* pool_Wr = (const float*)d_in[8];
  const float* pool_Wn = (const float*)d_in[9];
  const float* pool_b = (const float*)d_in[10];
  const float* lin1_W = (const float*)d_in[11];
  const float* lin1_b = (const float*)d_in[12];
  const float* lin2_W = (const float*)d_in[13];
  const float* lin2_b = (const float*)d_in[14];
  const float* head_W = (const float*)d_in[15];
  const float* head_b = (const float*)d_in[16];
  float* out = (float*)d_out;

  char* ws = (char*)d_ws;
  size_t off = 0;
  auto take = [&](size_t bytes) {
    void* p = ws + off;
    off += (bytes + 255) & ~(size_t)255;
    return p;
  };
  float* hX = (float*)take((size_t)NT0 * D * 4);
  float* hY = (float*)take((size_t)NT0 * D * 4);
  int* srcW = (int*)take((size_t)E_EDGES * 4);
  int* dstW = (int*)take((size_t)E_EDGES * 4);
  float* emW = (float*)take((size_t)E_EDGES * 4);
  float* deg = (float*)take((size_t)NT0 * 4);   // becomes dinv in place
  float* sc = (float*)take((size_t)NT0 * 4);
  float* tvec = (float*)take((size_t)NT0 * 4);
  int* newpos = (int*)take((size_t)NT0 * 4);
  int* kept = (int*)take((size_t)(NT0 / 2) * 4);
  float* gsum = (float*)take((size_t)B_GRAPHS * 2 * D * 4);
  __bf16* wpack = (__bf16*)take((size_t)3 * 16384 * 2);

  auto cdiv = [](int a, int b) { return (a + b - 1) / b; };
  const int TB = 256;

  pack_w_kernel<<<cdiv(3 * 16384, TB), TB, 0, stream>>>(conv_W, wpack);
  embed_kernel<<<cdiv(NT0 * D, TB), TB, 0, stream>>>(x, depth, emb_type, emb_depth, hX);
  init_edges_kernel<<<cdiv(E_EDGES, TB), TB, 0, stream>>>(edge_index, srcW, dstW, emW);
  fill_f32_kernel<<<cdiv(B_GRAPHS * 2 * D, TB), TB, 0, stream>>>(gsum, 0.0f, B_GRAPHS * 2 * D);

  float* cur = hX;
  float* alt = hY;
  int n = N0;
  for (int l = 0; l < 3; ++l) {
    const int nt = B_GRAPHS * n;
    const int k = n / 2;

    // hw = h @ W  (WMMA bf16, f32 accumulate)
    gemm_bf16_kernel<<<nt / 128, TB, 0, stream>>>(cur, wpack + (size_t)l * 16384, alt, nt);

    // degree with self loop, then dinv
    fill_f32_kernel<<<cdiv(nt, TB), TB, 0, stream>>>(deg, 1.0f, nt);
    deg_scatter_kernel<<<cdiv(E_EDGES, TB), TB, 0, stream>>>(dstW, emW, deg);
    rsqrt_kernel<<<cdiv(nt, TB), TB, 0, stream>>>(deg, nt);

    // agg (overwrites dead h buffer), then combine + relu in place
    fill_f32_kernel<<<cdiv(nt * D, TB), TB, 0, stream>>>(cur, 0.0f, nt * D);
    gcn_scatter_kernel<<<E_EDGES / 8, TB, 0, stream>>>(alt, srcW, dstW, emW, deg, cur);
    combine_kernel<<<cdiv(nt * D, TB), TB, 0, stream>>>(cur, alt, deg, conv_b + l * D, nt);

    // SAGPool score
    score_node_kernel<<<cdiv(nt * 32, TB), TB, 0, stream>>>(
        cur, pool_Wr + l * D, pool_Wn + l * D, pool_b + l, sc, tvec, nt);
    score_scatter_kernel<<<cdiv(E_EDGES, TB), TB, 0, stream>>>(srcW, dstW, emW, tvec, sc);

    // top-k selection + pooling + edge remap
    fill_i32_kernel<<<cdiv(nt, TB), TB, 0, stream>>>(newpos, -1, nt);
    topk_kernel<<<B_GRAPHS, TB, 0, stream>>>(sc, n, k, kept, newpos);
    pool_gather_kernel<<<cdiv(B_GRAPHS * k * D, TB), TB, 0, stream>>>(cur, kept, sc, alt,
                                                                      B_GRAPHS * k);
    remap_kernel<<<cdiv(E_EDGES, TB), TB, 0, stream>>>(newpos, srcW, dstW, emW);

    // readout accumulation (max | mean)
    readout_kernel<<<B_GRAPHS, TB, 0, stream>>>(alt, k, gsum);

    // ping-pong
    float* tmp = cur; cur = alt; alt = tmp;
    n = k;
  }

  mlp_kernel<<<B_GRAPHS, D, 0, stream>>>(gsum, lin1_W, lin1_b, lin2_W, lin2_b,
                                         head_W, head_b, out);
}
